// ColumnBlock_24335284699291
// MI455X (gfx1250) — compile-verified, matrix operand order checked by reading, not run on hardware
//
#include <hip/hip_runtime.h>
#include <hip/hip_bf16.h>

// ---------------------------------------------------------------------------
// ColumnBlock on MI455X (gfx1250): bf16 WMMA (16x16x32) with f32 accumulation.
// ---------------------------------------------------------------------------

typedef __bf16 bf16;
typedef __attribute__((ext_vector_type(16))) __bf16 v16bf;
typedef __attribute__((ext_vector_type(8)))  float  v8f;

static constexpr int kB  = 32;
static constexpr int kN  = 1024;
static constexpr int kC  = 384;
static constexpr int kD  = 48;
static constexpr int kDP = 64;        // D padded to a multiple of 32 (zeros)
static constexpr float kEPS = 1e-5f;

// K index inside a 16x32 bf16 fragment: element pair p (0..7), lane half hi.
__device__ __forceinline__ int frag_k(int p, int hi) {
    return 2 * p + ((p >= 4) ? 8 : 0) + (hi ? 8 : 0);
}

// A fragment (16 rows x 32 K) from row-major bf16, row stride `stride`.
// `src` points at (row0, k0). Lane 0..15 -> rows, halves split K.
__device__ __forceinline__ v16bf load_fragA(const bf16* __restrict__ src,
                                            long stride, int lane) {
    int r = lane & 15, hi = lane >> 4;
    v16bf f;
#pragma unroll
    for (int p = 0; p < 8; ++p) {
        int k = frag_k(p, hi);
        f[2 * p]     = src[(long)r * stride + k];
        f[2 * p + 1] = src[(long)r * stride + k + 1];
    }
    return f;
}

// B fragment from a transposed bf16 source: B[k][col] = src[col*stride + k].
// src points at (col0, k0). Pairs are contiguous along k -> b32 loads.
__device__ __forceinline__ v16bf load_fragB_T(const bf16* __restrict__ src,
                                              long stride, int lane) {
    int c = lane & 15, hi = lane >> 4;
    v16bf f;
#pragma unroll
    for (int p = 0; p < 8; ++p) {
        int k = frag_k(p, hi);
        f[2 * p]     = src[(long)c * stride + k];
        f[2 * p + 1] = src[(long)c * stride + k + 1];
    }
    return f;
}

// Same, but source is f32 weights converted to bf16 on the fly, with a K
// guard (kmax relative to src's k0) so padded K reads past a row are zero.
__device__ __forceinline__ v16bf load_fragB_Tf32(const float* __restrict__ src,
                                                 long stride, int lane, int kmax) {
    int c = lane & 15, hi = lane >> 4;
    v16bf f;
#pragma unroll
    for (int p = 0; p < 8; ++p) {
        int k = frag_k(p, hi);
        f[2 * p]     = (k     < kmax) ? (bf16)src[(long)c * stride + k]     : (bf16)0.f;
        f[2 * p + 1] = (k + 1 < kmax) ? (bf16)src[(long)c * stride + k + 1] : (bf16)0.f;
    }
    return f;
}

__device__ __forceinline__ v8f wmma_bf16(v16bf a, v16bf b, v8f c) {
    return __builtin_amdgcn_wmma_f32_16x16x32_bf16(
        /*neg_a=*/false, a, /*neg_b=*/false, b,
        /*c_mod=*/(short)0, c, /*reuse_a=*/false, /*reuse_b=*/false);
}

// Store a 16x16 f32 accumulator tile (ISA C/D layout) row-major.
__device__ __forceinline__ void store_tile(float* __restrict__ dst, long stride,
                                           int lane, v8f acc) {
    int c = lane & 15, hi = lane >> 4;
#pragma unroll
    for (int r = 0; r < 8; ++r)
        dst[(long)(r + 8 * hi) * stride + c] = acc[r];
}

// ---------------------------------------------------------------------------
// 1) LayerNorm rows (C=384) -> bf16 (+ optional transposed copy hT[b][c][n]).
//    One wave per (b,n) row.
// ---------------------------------------------------------------------------
__global__ void cb_ln_kernel(const float* __restrict__ x,
                             const float* __restrict__ w,
                             const float* __restrict__ b,
                             bf16* __restrict__ out,
                             bf16* __restrict__ outT /* nullable, [B,C,N] */) {
    int wave = (blockIdx.x * blockDim.x + threadIdx.x) >> 5;
    int lane = threadIdx.x & 31;
    int n  = wave % kN;
    int bi = wave / kN;
    const float* row = x + (long)wave * kC;
    float v[kC / 32];
    float s1 = 0.f, s2 = 0.f;
#pragma unroll
    for (int i = 0; i < kC / 32; ++i) {
        float t = row[lane + 32 * i];
        v[i] = t; s1 += t; s2 += t * t;
    }
#pragma unroll
    for (int off = 16; off; off >>= 1) {
        s1 += __shfl_xor(s1, off, 32);
        s2 += __shfl_xor(s2, off, 32);
    }
    float mu  = s1 * (1.f / kC);
    float var = s2 * (1.f / kC) - mu * mu;
    float rs  = rsqrtf(var + kEPS);
    const float* wr = w + (long)n * kC;
    const float* br = b + (long)n * kC;
    bf16* orow = out + (long)wave * kC;
    if (outT) {
        bf16* ot = outT + (long)bi * kC * kN + n;
#pragma unroll
        for (int i = 0; i < kC / 32; ++i) {
            int c = lane + 32 * i;
            bf16 t = (bf16)((v[i] - mu) * rs * wr[c] + br[c]);
            orow[c] = t;
            ot[(long)c * kN] = t;
        }
    } else {
#pragma unroll
        for (int i = 0; i < kC / 32; ++i) {
            int c = lane + 32 * i;
            orow[c] = (bf16)((v[i] - mu) * rs * wr[c] + br[c]);
        }
    }
}

// ---------------------------------------------------------------------------
// 2) Per-n Q/K projection: [32 x 384] (h rows over b) x wq[n]^T -> [32 x 48].
//    One wave per (which, n, dtile); weights converted f32->bf16 on the fly.
// ---------------------------------------------------------------------------
__global__ void cb_qk_kernel(const bf16* __restrict__ h,
                             const float* __restrict__ wq, const float* __restrict__ bq,
                             const float* __restrict__ wk, const float* __restrict__ bk,
                             bf16* __restrict__ qout, bf16* __restrict__ kout,
                             float scale) {
    int wave = (blockIdx.x * blockDim.x + threadIdx.x) >> 5;
    int lane = threadIdx.x & 31;
    int which = wave / (kN * 3);          // 0 = q, 1 = k
    int rem   = wave % (kN * 3);
    int n  = rem / 3;
    int dt = rem % 3;
    const float* w    = which ? wk : wq;
    const float* bias = which ? bk : bq;
    bf16* out         = which ? kout : qout;
    float scl         = which ? 1.f : scale;

    const bf16*  A0 = h + (long)n * kC;                 // row b, stride N*C
    const float* Wn = w + (long)n * kD * kC;            // [D, C] row-major
    int d0 = dt * 16;
    v8f acc0 = {}, acc1 = {};
    for (int k0 = 0; k0 < kC; k0 += 32) {
        v16bf a0 = load_fragA(A0 + k0, (long)kN * kC, lane);
        v16bf a1 = load_fragA(A0 + (long)16 * kN * kC + k0, (long)kN * kC, lane);
        v16bf bb = load_fragB_Tf32(Wn + (long)d0 * kC + k0, kC, lane, kC);
        acc0 = wmma_bf16(a0, bb, acc0);
        acc1 = wmma_bf16(a1, bb, acc1);
    }
    int col = lane & 15, hi = lane >> 4;
    int d = d0 + col;
    float bv = bias[(long)n * kD + d];
#pragma unroll
    for (int r = 0; r < 8; ++r) {
        int b0 = r + 8 * hi;
        out[((long)b0 * kN + n) * kDP + d]        = (bf16)(scl * (acc0[r] + bv));
        out[((long)(b0 + 16) * kN + n) * kDP + d] = (bf16)(scl * (acc1[r] + bv));
    }
    if (dt == 2) {  // zero the K-padding d = 48..63 (row b = lane)
#pragma unroll
        for (int d2 = kD; d2 < kDP; ++d2)
            out[((long)lane * kN + n) * kDP + d2] = (bf16)0.f;
    }
}

// ---------------------------------------------------------------------------
// 3) S = q . k^T per batch: wave per 16(n) x 64(m) strip, K = 64 (padded).
//    The q fragment is shared across 4 m-tiles -> 8 WMMAs per wave.
// ---------------------------------------------------------------------------
__global__ void cb_qkt_kernel(const bf16* __restrict__ q,
                              const bf16* __restrict__ k,
                              float* __restrict__ S) {
    int wave = (blockIdx.x * blockDim.x + threadIdx.x) >> 5;
    int lane = threadIdx.x & 31;
    int b   = wave / (64 * 16);
    int rem = wave % (64 * 16);
    int nt = rem / 16, mq = rem % 16;
    int m0 = mq * 64;
    const bf16* qb = q + (long)b * kN * kDP + (long)nt * 16 * kDP;
    const bf16* kb = k + (long)b * kN * kDP + (long)m0 * kDP;
    v8f acc[4] = {{}, {}, {}, {}};
#pragma unroll
    for (int k0 = 0; k0 < kDP; k0 += 32) {
        v16bf a = load_fragA(qb + k0, kDP, lane);
#pragma unroll
        for (int j = 0; j < 4; ++j) {
            v16bf bb = load_fragB_T(kb + (long)j * 16 * kDP + k0, kDP, lane);
            acc[j] = wmma_bf16(a, bb, acc[j]);
        }
    }
    float* out = S + (long)b * kN * kN + (long)nt * 16 * kN + m0;
#pragma unroll
    for (int j = 0; j < 4; ++j)
        store_tile(out + j * 16, kN, lane, acc[j]);
}

// ---------------------------------------------------------------------------
// 4) Row softmax over N=1024; f32 result in place (d_out) + bf16 copy.
// ---------------------------------------------------------------------------
__global__ void cb_softmax_kernel(float* __restrict__ S, bf16* __restrict__ Sbf) {
    int wave = (blockIdx.x * blockDim.x + threadIdx.x) >> 5;
    int lane = threadIdx.x & 31;
    float* row = S + (long)wave * kN;
    float v[32];
    float mx = -3.0e38f;
#pragma unroll
    for (int i = 0; i < 32; ++i) { v[i] = row[lane + 32 * i]; mx = fmaxf(mx, v[i]); }
#pragma unroll
    for (int off = 16; off; off >>= 1) mx = fmaxf(mx, __shfl_xor(mx, off, 32));
    float s = 0.f;
#pragma unroll
    for (int i = 0; i < 32; ++i) { v[i] = __expf(v[i] - mx); s += v[i]; }
#pragma unroll
    for (int off = 16; off; off >>= 1) s += __shfl_xor(s, off, 32);
    float inv = 1.f / s;
    bf16* brow = Sbf + (long)wave * kN;
#pragma unroll
    for (int i = 0; i < 32; ++i) {
        float t = v[i] * inv;
        row[lane + 32 * i]  = t;
        brow[lane + 32 * i] = (bf16)t;
    }
}

// ---------------------------------------------------------------------------
// 5) O = attn . h + x  (the 25.8 GF GEMM). Wave per 16(n) x 64(c) tile using
//    the transposed hT[b][c][m] so B fragments are contiguous b32 pair loads.
//    K = 1024 -> 32 steps, 4 WMMAs/step sharing one A fragment.
// ---------------------------------------------------------------------------
__global__ void cb_av_kernel(const bf16* __restrict__ attn,
                             const bf16* __restrict__ hT,
                             const float* __restrict__ x,
                             float* __restrict__ xout) {
    int wave = (blockIdx.x * blockDim.x + threadIdx.x) >> 5;
    int lane = threadIdx.x & 31;
    int b   = wave / (64 * 6);
    int rem = wave % (64 * 6);
    int nt = rem / 6, cq = rem % 6;
    int c0 = cq * 64;
    const bf16* A0 = attn + (long)b * kN * kN + (long)nt * 16 * kN;
    const bf16* B0 = hT + (long)b * kC * kN + (long)c0 * kN;   // [c][m]
    v8f acc[4] = {{}, {}, {}, {}};
    for (int k0 = 0; k0 < kN; k0 += 32) {
        __builtin_prefetch(A0 + k0 + 128, 0, 0);   // global_prefetch_b8
        v16bf a = load_fragA(A0 + k0, kN, lane);
#pragma unroll
        for (int j = 0; j < 4; ++j) {
            v16bf bb = load_fragB_T(B0 + (long)j * 16 * kN + k0, kN, lane);
            acc[j] = wmma_bf16(a, bb, acc[j]);
        }
    }
    int col = lane & 15, hi = lane >> 4;
#pragma unroll
    for (int r = 0; r < 8; ++r) {
        int nrow = nt * 16 + r + 8 * hi;
        long base = ((long)b * kN + nrow) * kC + c0 + col;
#pragma unroll
        for (int j = 0; j < 4; ++j)
            xout[base + 16 * j] = acc[j][r] + x[base + 16 * j];
    }
}

// ---------------------------------------------------------------------------
// 6) MLP up-projection + exact GELU: like cb_qk but single matrix.
// ---------------------------------------------------------------------------
__global__ void cb_mlp1_kernel(const bf16* __restrict__ h2,
                               const float* __restrict__ w1,
                               const float* __restrict__ b1,
                               bf16* __restrict__ m) {
    int wave = (blockIdx.x * blockDim.x + threadIdx.x) >> 5;
    int lane = threadIdx.x & 31;
    int n = wave / 3, dt = wave % 3;
    const bf16*  A0 = h2 + (long)n * kC;
    const float* Wn = w1 + (long)n * kD * kC;
    int d0 = dt * 16;
    v8f acc0 = {}, acc1 = {};
    for (int k0 = 0; k0 < kC; k0 += 32) {
        v16bf a0 = load_fragA(A0 + k0, (long)kN * kC, lane);
        v16bf a1 = load_fragA(A0 + (long)16 * kN * kC + k0, (long)kN * kC, lane);
        v16bf bb = load_fragB_Tf32(Wn + (long)d0 * kC + k0, kC, lane, kC);
        acc0 = wmma_bf16(a0, bb, acc0);
        acc1 = wmma_bf16(a1, bb, acc1);
    }
    int col = lane & 15, hi = lane >> 4;
    int d = d0 + col;
    float bv = b1[(long)n * kD + d];
#pragma unroll
    for (int r = 0; r < 8; ++r) {
        int b0 = r + 8 * hi;
        float t0 = acc0[r] + bv;
        float t1 = acc1[r] + bv;
        t0 = 0.5f * t0 * (1.f + erff(t0 * 0.70710678118f));
        t1 = 0.5f * t1 * (1.f + erff(t1 * 0.70710678118f));
        m[((long)b0 * kN + n) * kDP + d]        = (bf16)t0;
        m[((long)(b0 + 16) * kN + n) * kDP + d] = (bf16)t1;
    }
    if (dt == 2) {
#pragma unroll
        for (int d2 = kD; d2 < kDP; ++d2)
            m[((long)lane * kN + n) * kDP + d2] = (bf16)0.f;
    }
}

// ---------------------------------------------------------------------------
// 7) MLP down-projection + residual, K guarded at D=48 (w2 rows are length 48).
// ---------------------------------------------------------------------------
__global__ void cb_mlp2_kernel(const bf16* __restrict__ m,
                               const float* __restrict__ w2,
                               const float* __restrict__ b2,
                               float* __restrict__ x /* in-out */) {
    int wave = (blockIdx.x * blockDim.x + threadIdx.x) >> 5;
    int lane = threadIdx.x & 31;
    int n = wave / 24, ct = wave % 24;
    const bf16*  A0 = m + (long)n * kDP;                        // row b, stride N*DP
    const float* Wn = w2 + (long)n * kC * kD + (long)ct * 16 * kD;
    v8f acc0 = {}, acc1 = {};
#pragma unroll
    for (int k0 = 0; k0 < kDP; k0 += 32) {
        v16bf a0 = load_fragA(A0 + k0, (long)kN * kDP, lane);
        v16bf a1 = load_fragA(A0 + (long)16 * kN * kDP + k0, (long)kN * kDP, lane);
        v16bf bb = load_fragB_Tf32(Wn + k0, kD, lane, kD - k0);
        acc0 = wmma_bf16(a0, bb, acc0);
        acc1 = wmma_bf16(a1, bb, acc1);
    }
    int col = lane & 15, hi = lane >> 4;
    int c = ct * 16 + col;
    float bv = b2[(long)n * kC + c];
#pragma unroll
    for (int r = 0; r < 8; ++r) {
        int b0 = r + 8 * hi;
        long i0 = ((long)b0 * kN + n) * kC + c;
        long i1 = ((long)(b0 + 16) * kN + n) * kC + c;
        x[i0] = x[i0] + acc0[r] + bv;
        x[i1] = x[i1] + acc1[r] + bv;
    }
}

// ---------------------------------------------------------------------------
extern "C" void kernel_launch(void* const* d_in, const int* in_sizes, int n_in,
                              void* d_out, int out_size, void* d_ws, size_t ws_size,
                              hipStream_t stream) {
    const float* x   = (const float*)d_in[0];
    const float* n1w = (const float*)d_in[1];
    const float* n1b = (const float*)d_in[2];
    const float* wq  = (const float*)d_in[3];
    const float* bq  = (const float*)d_in[4];
    const float* wk  = (const float*)d_in[5];
    const float* bk  = (const float*)d_in[6];
    const float* n2w = (const float*)d_in[7];
    const float* n2b = (const float*)d_in[8];
    const float* w1  = (const float*)d_in[9];
    const float* b1  = (const float*)d_in[10];
    const float* w2  = (const float*)d_in[11];
    const float* b2  = (const float*)d_in[12];

    float* xout     = (float*)d_out;                       // [B,N,C]
    float* attn_out = xout + (long)kB * kN * kC;           // [B,N,N]

    char* ws = (char*)d_ws;
    auto take = [&](size_t bytes) {
        char* p = ws;
        ws += (bytes + 255) & ~(size_t)255;
        return p;
    };
    bf16* h_bf    = (bf16*)take((size_t)kB * kN * kC * sizeof(bf16));
    bf16* hT_bf   = (bf16*)take((size_t)kB * kC * kN * sizeof(bf16));
    bf16* h2_bf   = (bf16*)take((size_t)kB * kN * kC * sizeof(bf16));
    bf16* q_bf    = (bf16*)take((size_t)kB * kN * kDP * sizeof(bf16));
    bf16* k_bf    = (bf16*)take((size_t)kB * kN * kDP * sizeof(bf16));
    bf16* m_bf    = (bf16*)take((size_t)kB * kN * kDP * sizeof(bf16));
    bf16* attn_bf = (bf16*)take((size_t)kB * kN * kN * sizeof(bf16));

    const float scale = 1.f / sqrtf((float)kC);

    // 1) h = LN1(x)  (+ transposed copy for the AV GEMM's B side)
    cb_ln_kernel<<<(kB * kN) / 8, 256, 0, stream>>>(x, n1w, n1b, h_bf, hT_bf);
    // 2) q, k  (waves: 2 * N * 3 = 6144 -> 768 blocks)
    cb_qk_kernel<<<768, 256, 0, stream>>>(h_bf, wq, bq, wk, bk, q_bf, k_bf, scale);
    // 3) S = q k^T  (waves: 32*64*16 = 32768 -> 4096 blocks)
    cb_qkt_kernel<<<4096, 256, 0, stream>>>(q_bf, k_bf, attn_out);
    // 4) softmax rows (32768 rows -> 4096 blocks)
    cb_softmax_kernel<<<4096, 256, 0, stream>>>(attn_out, attn_bf);
    // 5) x2 = x + attn.h  (waves: 32*64*6 = 12288 -> 1536 blocks)
    cb_av_kernel<<<1536, 256, 0, stream>>>(attn_bf, hT_bf, x, xout);
    // 6) h2 = LN2(x2)
    cb_ln_kernel<<<(kB * kN) / 8, 256, 0, stream>>>(xout, n2w, n2b, h2_bf, nullptr);
    // 7) m = gelu(h2 w1^T + b1)  (waves: N*3 = 3072 -> 384 blocks)
    cb_mlp1_kernel<<<384, 256, 0, stream>>>(h2_bf, w1, b1, m_bf);
    // 8) x += m w2^T + b2  (waves: N*24 = 24576 -> 3072 blocks)
    cb_mlp2_kernel<<<3072, 256, 0, stream>>>(m_bf, w2, b2, xout);
}
